// GraphTemporalGNN_9740985828027
// MI455X (gfx1250) — matrix operand order repaired and checked by reading, not compiled, not run on hardware
//
#include <hip/hip_runtime.h>
#include <hip/hip_bf16.h>
#include <stddef.h>
#include <stdint.h>

// ---------------------------------------------------------------------------
// GraphTemporalGNN for MI455X (gfx1250, wave32).
// Big GEMMs (h @ W, 208896x128x128) run on v_wmma_f32_16x16x32_bf16.
// Weight staged per-block into LDS via global_load_async_to_lds_b128, then
// register-resident per wave; waves are persistent over 512-row M-strips.
// ---------------------------------------------------------------------------

#define B_      32
#define T_      96
#define NN_     68
#define NNODES  (B_ * T_ * NN_)        // 208896
#define NEDGES  (NNODES * 8)           // 1671168
#define HID     128
#define NF      ((size_t)NNODES * HID) // 26,738,688 elements
#define STRIP   8                      // 64-row chunks per block (512 rows)

typedef __attribute__((ext_vector_type(16))) __bf16        v16bf;
typedef __attribute__((ext_vector_type(8)))  float         v8f;
typedef __attribute__((ext_vector_type(8)))  unsigned int  v8u;
typedef __attribute__((ext_vector_type(4)))  unsigned int  v4u;

__device__ __forceinline__ unsigned short f2bf(float f) {
    unsigned int u = __float_as_uint(f);
    u += 0x7FFFu + ((u >> 16) & 1u);      // round-to-nearest-even
    return (unsigned short)(u >> 16);
}

__device__ __forceinline__ void atomic_add_f32(float* p, float v) {
    __hip_atomic_fetch_add(p, v, __ATOMIC_RELAXED, __HIP_MEMORY_SCOPE_AGENT);
}

// ---------------------------------------------------------------------------
// Small prep kernels
// ---------------------------------------------------------------------------

// Wt[n*K + k] = bf16(W[k*N + n]); launched with grid=N, block=K (K,N = 128)
__global__ void k_w_to_bf16_t(const float* __restrict__ W,
                              unsigned short* __restrict__ Wt, int K, int N) {
    int n = blockIdx.x, k = threadIdx.x;
    Wt[(size_t)n * K + k] = f2bf(W[(size_t)k * N + n]);
}

__global__ void k_deg_init(float* __restrict__ deg, int n) {
    int i = blockIdx.x * blockDim.x + threadIdx.x;
    if (i < n) deg[i] = 1.0f;
}

__global__ void k_deg_scatter(const int* __restrict__ dst, float* __restrict__ deg, int n) {
    int i = blockIdx.x * blockDim.x + threadIdx.x;
    if (i < n) atomic_add_f32(&deg[dst[i]], 1.0f);
}

__global__ void k_dinv(const float* __restrict__ deg, float* __restrict__ dis, int n) {
    int i = blockIdx.x * blockDim.x + threadIdx.x;
    if (i < n) dis[i] = rsqrtf(deg[i]);
}

// ---------------------------------------------------------------------------
// GCN0: hw = x @ W0  (K = 3; memory-bound, plain VALU)
// grid = NNODES, block = 128
// ---------------------------------------------------------------------------
__global__ void k_gcn0(const float* __restrict__ x, const float* __restrict__ W0,
                       float* __restrict__ hw) {
    int i = blockIdx.x, f = threadIdx.x;
    float x0 = x[(size_t)i * 3 + 0];
    float x1 = x[(size_t)i * 3 + 1];
    float x2 = x[(size_t)i * 3 + 2];
    hw[(size_t)i * HID + f] = x0 * W0[f] + x1 * W0[HID + f] + x2 * W0[2 * HID + f];
}

// out = hw * self_norm + bias   (self_norm = dis^2 = 1/deg)
// grid = NNODES, block = 128
__global__ void k_init_out(const float* __restrict__ hw, const float* __restrict__ dis,
                           const float* __restrict__ bias, float* __restrict__ out) {
    int i = blockIdx.x, f = threadIdx.x;
    float sn = dis[i] * dis[i];
    out[(size_t)i * HID + f] = hw[(size_t)i * HID + f] * sn + bias[f];
}

// out[dst] += hw[src] * (dis[src]*dis[dst])  -- grid = NEDGES, block = 128
__global__ void k_edge_scatter(const int* __restrict__ src, const int* __restrict__ dst,
                               const float* __restrict__ dis, const float* __restrict__ hw,
                               float* __restrict__ out) {
    int e = blockIdx.x, f = threadIdx.x;
    int s = src[e], d = dst[e];
    float en = dis[s] * dis[d];
    atomic_add_f32(&out[(size_t)d * HID + f], hw[(size_t)s * HID + f] * en);
}

// h = relu(h) in place; also emit bf16 copy for next WMMA GEMM
__global__ void k_relu_bf16(float* __restrict__ h, unsigned short* __restrict__ hb, size_t n) {
    size_t i = (size_t)blockIdx.x * blockDim.x + threadIdx.x;
    if (i < n) {
        float v = fmaxf(h[i], 0.0f);
        h[i] = v;
        hb[i] = f2bf(v);
    }
}

// ---------------------------------------------------------------------------
// WMMA GEMM: C[M x 128] = A[M x 128](bf16, row-major) @ B[128 x 128]
// B supplied pre-transposed bf16: Bt[n*128 + k].
//  - Stage B once per block into LDS with async copies (ASYNCcnt).
//  - Each wave preloads its 16 B fragments (64 cols x 128 k) into 128 VGPRs.
//  - Wave persists over an 8-tile M-strip: per 16x64 tile only 8 NT A-loads
//    feed 16 back-to-back WMMAs; no LDS traffic in the steady state.
//  - FUSED template param folds bias+relu at compile time; bias values are
//    hoisted into registers outside the persistent loop.
// Block = 256 threads (8 waves): wave&3 -> M subtile, wave>>2 -> 64-col group.
// grid = M / 512 = 408.
//
// Fragment layouts per CDNA5 ISA §7.12.2:
//   A 16x32: lane m(<16): K = {0..7, 16..23}; lane m+16: K = {8..15, 24..31}
//   B 32x16: lane n(<16): col n, K = {0..7, 16..23}; lane n+16: col n, K = {8..15, 24..31}
//   C/D:     VGPR r: row = r + 8*(lane>=16), col = lane%16
// ---------------------------------------------------------------------------
template <bool FUSED>
__global__ __launch_bounds__(256) void k_wmma_gemm128(
    const unsigned short* __restrict__ Abf,
    const unsigned short* __restrict__ Bt,
    float* __restrict__ C,
    const float* __restrict__ bias)   // read only when FUSED
{
    __shared__ __align__(16) unsigned short Bs[HID * HID];   // 32 KB weight tile

    const int tid      = threadIdx.x;
    const int wave     = tid >> 5;
    const int lane     = tid & 31;
    const int laneHalf = lane >> 4;   // 0 | 1
    const int lmod     = lane & 15;
    const int nBase    = (wave >> 2) * 64;

    // ---- Stage the full 128x128 bf16 weight into LDS with async copies ----
    // 2048 x 16B chunks; 256 threads x 8 chunks each. Per ISA §10 the generic
    // address of LDS has the raw LDS byte offset in addr[31:0].
    {
        const unsigned ldsBase = (unsigned)(uintptr_t)(&Bs[0]);
#pragma unroll
        for (int j = 0; j < 8; ++j) {
            const int c = tid + j * 256;
            const unsigned la = ldsBase + ((unsigned)c << 4);
            const unsigned long long ga =
                (unsigned long long)(uintptr_t)(Bt + ((size_t)c << 3));
            asm volatile("global_load_async_to_lds_b128 %0, %1, off"
                         :: "v"(la), "v"(ga) : "memory");
        }
        asm volatile("s_wait_asynccnt 0x0" ::: "memory");
        __syncthreads();
    }

    // ---- Preload this wave's 16 B fragments into registers (128 VGPRs) ----
    v16bf bfr[4][4];                   // [kstep][n-subtile]
#pragma unroll
    for (int ks = 0; ks < 4; ++ks) {
#pragma unroll
        for (int s = 0; s < 4; ++s) {
            const int boff = (nBase + s * 16 + lmod) * HID + ks * 32 + laneHalf * 8;
            v4u bl = *(const v4u*)&Bs[boff];        // K = +0 .. +7
            v4u bh = *(const v4u*)&Bs[boff + 16];   // K = +16 .. +23
            v8u bu = {bl.x, bl.y, bl.z, bl.w, bh.x, bh.y, bh.z, bh.w};
            bfr[ks][s] = __builtin_bit_cast(v16bf, bu);
        }
    }

    // ---- Hoist per-wave bias values (loop-invariant) ----
    float bv[4] = {0.0f, 0.0f, 0.0f, 0.0f};
    if (FUSED) {
#pragma unroll
        for (int s = 0; s < 4; ++s) bv[s] = bias[nBase + s * 16 + lmod];
    }

    // ---- Persistent M-strip loop: 8 tiles of 16 rows per wave ----
    for (int mt = 0; mt < STRIP; ++mt) {
        const int mBase = (blockIdx.x * STRIP + mt) * 64 + (wave & 3) * 16;
        const unsigned short* arow = Abf + (size_t)(mBase + lmod) * HID;

        // All 4 K-step A fragments up-front (clauses of NT b128 loads).
        v16bf afr[4];
#pragma unroll
        for (int ks = 0; ks < 4; ++ks) {
            const int ka = ks * 32 + laneHalf * 8;
            v4u al = __builtin_nontemporal_load((const v4u*)(arow + ka));       // K=ka..ka+7
            v4u ah = __builtin_nontemporal_load((const v4u*)(arow + ka + 16));  // K=ka+16..ka+23
            v8u au = {al.x, al.y, al.z, al.w, ah.x, ah.y, ah.z, ah.w};
            afr[ks] = __builtin_bit_cast(v16bf, au);
        }

        v8f acc[4] = {};
#pragma unroll
        for (int ks = 0; ks < 4; ++ks) {
#pragma unroll
            for (int s = 0; s < 4; ++s) {
                acc[s] = __builtin_amdgcn_wmma_f32_16x16x32_bf16(
                    false, afr[ks], false, bfr[ks][s], (short)0, acc[s], false, false);
            }
        }

#pragma unroll
        for (int s = 0; s < 4; ++s) {
            const int col = nBase + s * 16 + lmod;
#pragma unroll
            for (int r = 0; r < 8; ++r) {
                const int m = mBase + r + laneHalf * 8;
                float v;
                if (FUSED) v = fmaxf(acc[s][r] + bv[s], 0.0f);
                else       v = acc[s][r];
                C[(size_t)m * HID + col] = v;
            }
        }
    }
}

// ---------------------------------------------------------------------------
// Frame pooling: frames[bt, f] = mean_n h[(bt*68 + n), f]
// grid = B*T = 3072, block = 128
// ---------------------------------------------------------------------------
__global__ void k_pool(const float* __restrict__ h, float* __restrict__ frames) {
    int bt = blockIdx.x, f = threadIdx.x;
    const float* p = h + (size_t)bt * NN_ * HID + f;
    float s = 0.0f;
    for (int n = 0; n < NN_; ++n) s += p[(size_t)n * HID];
    frames[(size_t)bt * HID + f] = s * (1.0f / (float)NN_);
}

// ---------------------------------------------------------------------------
// GRU: batch-parallel (32 blocks), 128 threads = hidden dims, scan T=96 steps.
// ---------------------------------------------------------------------------
__global__ void k_gru(const float* __restrict__ frames,
                      const float* __restrict__ Wih, const float* __restrict__ Whh,
                      const float* __restrict__ bih, const float* __restrict__ bhh,
                      float* __restrict__ hT) {
    __shared__ float xs[HID];
    __shared__ float hs[HID];
    const int b = blockIdx.x, d = threadIdx.x;
    hs[d] = 0.0f;
    __syncthreads();

    const float* wr = Wih + (size_t)(0 * HID + d) * HID;
    const float* wz = Wih + (size_t)(1 * HID + d) * HID;
    const float* wn = Wih + (size_t)(2 * HID + d) * HID;
    const float* ur = Whh + (size_t)(0 * HID + d) * HID;
    const float* uz = Whh + (size_t)(1 * HID + d) * HID;
    const float* un = Whh + (size_t)(2 * HID + d) * HID;

    for (int t = 0; t < T_; ++t) {
        xs[d] = frames[((size_t)b * T_ + t) * HID + d];
        __syncthreads();
        float ir = bih[d], iz = bih[HID + d], in = bih[2 * HID + d];
        float hr = bhh[d], hz = bhh[HID + d], hn = bhh[2 * HID + d];
        for (int k = 0; k < HID; ++k) {
            float xv = xs[k], hv = hs[k];
            ir += wr[k] * xv;  iz += wz[k] * xv;  in += wn[k] * xv;
            hr += ur[k] * hv;  hz += uz[k] * hv;  hn += un[k] * hv;
        }
        float r  = 1.0f / (1.0f + __expf(-(ir + hr)));
        float z  = 1.0f / (1.0f + __expf(-(iz + hz)));
        float nn = tanhf(in + r * hn);
        float hnew = (1.0f - z) * nn + z * hs[d];
        __syncthreads();
        hs[d] = hnew;
        __syncthreads();
    }
    hT[(size_t)b * HID + d] = hs[d];
}

// ---------------------------------------------------------------------------
// Classifier: logits = relu(hT @ Wc1 + bc1) @ Wc2 + bc2. One block of 512.
// ---------------------------------------------------------------------------
__global__ void k_cls(const float* __restrict__ hT,
                      const float* __restrict__ Wc1, const float* __restrict__ bc1,
                      const float* __restrict__ Wc2, const float* __restrict__ bc2,
                      float* __restrict__ out) {
    __shared__ float hid[B_ * 64];
    const int tid = threadIdx.x;
    for (int i = tid; i < B_ * 64; i += 512) {
        int b = i >> 6, c = i & 63;
        float acc = bc1[c];
        for (int k = 0; k < HID; ++k) acc += hT[(size_t)b * HID + k] * Wc1[(size_t)k * 64 + c];
        hid[i] = fmaxf(acc, 0.0f);
    }
    __syncthreads();
    for (int i = tid; i < B_ * 10; i += 512) {
        int b = i / 10, c = i % 10;
        float acc = bc2[c];
        for (int k = 0; k < 64; ++k) acc += hid[b * 64 + k] * Wc2[k * 10 + c];
        out[i] = acc;
    }
}

// ---------------------------------------------------------------------------
// Host launcher
// ---------------------------------------------------------------------------
extern "C" void kernel_launch(void* const* d_in, const int* in_sizes, int n_in,
                              void* d_out, int out_size, void* d_ws, size_t ws_size,
                              hipStream_t stream) {
    const float* x   = (const float*)d_in[0];
    const int*   ei  = (const int*)d_in[1];
    const int*   src = ei;
    const int*   dst = ei + NEDGES;
    const float* W0  = (const float*)d_in[2];
    const float* b0  = (const float*)d_in[3];
    const float* W1  = (const float*)d_in[4];
    const float* b1  = (const float*)d_in[5];
    const float* Wm  = (const float*)d_in[6];
    const float* bm  = (const float*)d_in[7];
    const float* Wih = (const float*)d_in[8];
    const float* Whh = (const float*)d_in[9];
    const float* bih = (const float*)d_in[10];
    const float* bhh = (const float*)d_in[11];
    const float* Wc1 = (const float*)d_in[12];
    const float* bc1 = (const float*)d_in[13];
    const float* Wc2 = (const float*)d_in[14];
    const float* bc2 = (const float*)d_in[15];
    float* out = (float*)d_out;

    // Workspace carve (~271 MB total)
    char* base = (char*)d_ws;
    size_t off = 0;
    auto carve = [&](size_t bytes) -> void* {
        void* p = base + off;
        off = (off + bytes + 255) & ~(size_t)255;
        return p;
    };
    float*          deg    = (float*)carve((size_t)NNODES * 4);
    float*          dis    = (float*)carve((size_t)NNODES * 4);
    float*          bufH   = (float*)carve(NF * 4);            // h / out (in-place)
    float*          bufW   = (float*)carve(NF * 4);            // hw = h @ W
    unsigned short* hb     = (unsigned short*)carve(NF * 2);   // bf16 activations
    unsigned short* Wt1    = (unsigned short*)carve((size_t)HID * HID * 2);
    unsigned short* Wtm    = (unsigned short*)carve((size_t)HID * HID * 2);
    float*          frames = (float*)carve((size_t)B_ * T_ * HID * 4);
    float*          hT     = (float*)carve((size_t)B_ * HID * 4);
    (void)ws_size; (void)in_sizes; (void)n_in; (void)out_size;

    const int tn = 256;
    const int gbN = (NNODES + tn - 1) / tn;
    const int gbE = (NEDGES + tn - 1) / tn;
    const int gbF = (int)(NF / tn);            // exact: 104448
    const int gGemm = NNODES / (64 * STRIP);   // 408

    // Weight prep (transposed bf16 copies for WMMA B operands)
    k_w_to_bf16_t<<<dim3(HID), dim3(HID), 0, stream>>>(W1, Wt1, HID, HID);
    k_w_to_bf16_t<<<dim3(HID), dim3(HID), 0, stream>>>(Wm, Wtm, HID, HID);

    // Degrees and norms
    k_deg_init<<<dim3(gbN), dim3(tn), 0, stream>>>(deg, NNODES);
    k_deg_scatter<<<dim3(gbE), dim3(tn), 0, stream>>>(dst, deg, NEDGES);
    k_dinv<<<dim3(gbN), dim3(tn), 0, stream>>>(deg, dis, NNODES);

    // ---- GCN layer 0 (K=3 scalar matmul) ----
    k_gcn0<<<dim3(NNODES), dim3(HID), 0, stream>>>(x, W0, bufW);
    k_init_out<<<dim3(NNODES), dim3(HID), 0, stream>>>(bufW, dis, b0, bufH);
    k_edge_scatter<<<dim3(NEDGES), dim3(HID), 0, stream>>>(src, dst, dis, bufW, bufH);
    k_relu_bf16<<<dim3(gbF), dim3(tn), 0, stream>>>(bufH, hb, NF);

    // ---- GCN layer 1 (WMMA bf16 GEMM, plain) ----
    k_wmma_gemm128<false><<<dim3(gGemm), dim3(256), 0, stream>>>(hb, Wt1, bufW, nullptr);
    k_init_out<<<dim3(NNODES), dim3(HID), 0, stream>>>(bufW, dis, b1, bufH);
    k_edge_scatter<<<dim3(NEDGES), dim3(HID), 0, stream>>>(src, dst, dis, bufW, bufH);
    k_relu_bf16<<<dim3(gbF), dim3(tn), 0, stream>>>(bufH, hb, NF);

    // ---- Node MLP (WMMA bf16 GEMM, fused bias + relu) ----
    k_wmma_gemm128<true><<<dim3(gGemm), dim3(256), 0, stream>>>(hb, Wtm, bufH, bm);

    // ---- Frame pooling, GRU scan, classifier ----
    k_pool<<<dim3(B_ * T_), dim3(HID), 0, stream>>>(bufH, frames);
    k_gru<<<dim3(B_), dim3(HID), 0, stream>>>(frames, Wih, Whh, bih, bhh, hT);
    k_cls<<<dim3(1), dim3(512), 0, stream>>>(hT, Wc1, bc1, Wc2, bc2, out);
}
